// LocationAwareAttention_10075993276418
// MI455X (gfx1250) — matrix operand order repaired and checked by reading, not compiled, not run on hardware
//
#include <hip/hip_runtime.h>
#include <math.h>

// ---------------------------------------------------------------------------
// LocationAwareAttention on gfx1250 (MI455X)
// B=32, KLEN=2048, DEC=1024, ATT=512
// Dominant GEMM (65536x512x1024) via v_wmma_f32_16x16x32_bf16.
// Wk pre-converted to bf16 once; B tiles staged global->LDS with
// global_load_async_to_lds_b128 (ASYNCcnt), double-buffered so copies for
// chunk i+1 overlap the WMMAs of chunk i. A tile pipelined through registers.
// ---------------------------------------------------------------------------

#define B_   32
#define KLEN 2048
#define DEC  1024
#define ATT  512

typedef __bf16 v16bf __attribute__((ext_vector_type(16)));
typedef __bf16 v4bf  __attribute__((ext_vector_type(4)));
typedef float  v8f   __attribute__((ext_vector_type(8)));

__device__ __forceinline__ __bf16 f2bf(float f) {
  unsigned u = __builtin_bit_cast(unsigned, f);
  unsigned r = u + 0x7FFFu + ((u >> 16) & 1u);   // round-to-nearest-even
  unsigned short h = (unsigned short)(r >> 16);
  return __builtin_bit_cast(__bf16, h);
}

__device__ __forceinline__ v4bf f2bf4(float4 v) {
  v4bf o;
  o[0] = f2bf(v.x); o[1] = f2bf(v.y); o[2] = f2bf(v.z); o[3] = f2bf(v.w);
  return o;
}

__device__ __forceinline__ float fast_tanh(float x) {
  const float e = __expf(2.f * x);     // v_exp_f32
  return 1.f - 2.f / (e + 1.f);        // exact +/-1 limits, NaN propagates
}

// ------------------------------ kernel 0 -----------------------------------
// One-shot Wk f32 -> bf16, row-major [a][d] (1 MB, stays hot in 192MB L2).
__global__ __launch_bounds__(256) void convert_wk_kernel(
    const float* __restrict__ Wk, __bf16* __restrict__ out) {
  const int i = (blockIdx.x * 256 + threadIdx.x) * 4;
  *(v4bf*)(out + i) = f2bf4(*(const float4*)(Wk + i));
}

// ------------------------------ kernel 1 -----------------------------------
// qb[b*ATT+a] = dot(query[b,:], Wq[a,:]) + bias[a]
__global__ __launch_bounds__(256) void qproj_kernel(
    const float* __restrict__ q, const float* __restrict__ Wq,
    const float* __restrict__ bias, float* __restrict__ qb) {
  const int lane = threadIdx.x & 31;
  const int wave = threadIdx.x >> 5;
  const int idx  = blockIdx.x * 8 + wave;       // 0 .. B_*ATT-1
  const int b = idx >> 9;
  const int a = idx & 511;
  const float* qp = q  + (size_t)b * DEC;
  const float* wp = Wq + (size_t)a * DEC;
  float s = 0.f;
#pragma unroll 4
  for (int i = lane; i < DEC; i += 32) s += qp[i] * wp[i];
  s += __shfl_xor(s, 1);  s += __shfl_xor(s, 2);
  s += __shfl_xor(s, 4);  s += __shfl_xor(s, 8);  s += __shfl_xor(s, 16);
  if (lane == 0) qb[idx] = s + bias[a];
}

// ------------------------------ kernel 2 -----------------------------------
// 32-row (b,k) tile x all 512 ATT columns; k_proj via bf16 WMMA; fused
// epilogue tanh(kp + qb + conv(loc) + conv_b) . fc_w -> scores[b,k].
__global__ __launch_bounds__(256) void scores_wmma_kernel(
    const float* __restrict__ keyf, const __bf16* __restrict__ WkBf,
    const float* __restrict__ la,   const float* __restrict__ qb,
    const float* __restrict__ conv_w, const float* __restrict__ conv_b,
    const float* __restrict__ fc_w,  const float* __restrict__ fc_b,
    float* __restrict__ scores) {
  __shared__ __attribute__((aligned(16))) __bf16 lA[2][32][40];   // [buf][m][d]
  __shared__ __attribute__((aligned(16))) __bf16 lB[2][ATT][40];  // [buf][a][d]
  __shared__ float sc[32];

  const int t     = threadIdx.x;
  const int lane  = t & 31;
  const int wave  = t >> 5;
  const int waveM = wave >> 2;       // 0..1  (16-row slab)
  const int waveN = wave & 3;        // 0..3  (128-col slab)
  const int m0    = blockIdx.x * 32; // global row base (b*KLEN + k)
  const int b     = m0 >> 11;
  const int k0    = m0 & (KLEN - 1);

  if (t < 32) sc[t] = 0.f;

  v8f acc[8];
#pragma unroll
  for (int i = 0; i < 8; ++i)
#pragma unroll
    for (int j = 0; j < 8; ++j) acc[i][j] = 0.f;

  const int aRow = t >> 3;   // 0..31 : A-tile row staged by this thread
  const int dv   = t & 7;    // float4 group within 32-wide d-chunk
  const int aB   = t >> 2;   // B staging: base row
  const int segB = t & 3;    // 16B segment (8 bf16) within 64B row chunk

  // issue 8 async global->LDS copies (per wave) for one 512x32 bf16 B tile
  auto issueB = [&](int buf, int d0) {
#pragma unroll
    for (int j = 0; j < 8; ++j) {
      const int a = aB + j * 64;                      // 0..511
      const unsigned ldsoff =
          (unsigned)(size_t)(const void*)&lB[buf][a][segB * 8];
      const unsigned long long gaddr =
          (unsigned long long)(size_t)(const void*)(WkBf + (size_t)a * DEC + d0 + segB * 8);
      asm volatile("global_load_async_to_lds_b128 %0, %1, off"
                   :: "v"(ldsoff), "v"(gaddr) : "memory");
    }
  };

  // ---------------- pipelined main loop ----------------
  float4 areg = *(const float4*)(keyf + (size_t)(m0 + aRow) * DEC + dv * 4);
  issueB(0, 0);
  *(v4bf*)&lA[0][aRow][dv * 4] = f2bf4(areg);

  const int kb = ((lane >> 4) & 1) << 4;             // B frag K base kb/2 = 0 / 8 (table order)
  const int nL = (waveN << 7) + (lane & 15);         // N = lane&15

  for (int it = 0; it < 32; ++it) {
    const int d0  = it << 5;
    const int cur = it & 1;
    const int nxt = cur ^ 1;

    if (it < 31) {
      issueB(nxt, d0 + 32);                          // overlap with compute
      areg = *(const float4*)(keyf + (size_t)(m0 + aRow) * DEC + d0 + 32 + dv * 4);
      asm volatile("s_wait_asynccnt 0x8" ::: "memory");  // drain older tile only
    } else {
      asm volatile("s_wait_asynccnt 0x0" ::: "memory");
    }
    __syncthreads();   // buffer `cur` fully staged across all waves

    // --- A fragment: 16x32 bf16. lane = M (mod 16); K split lo/hi half ---
    v16bf afrag;
    {
      const int row = (waveM << 4) + (lane & 15);
      const int kh  = ((lane >> 4) & 1) << 3;        // +0 or +8
#pragma unroll
      for (int v = 0; v < 8; ++v) {
        const int kbb = ((v < 4) ? (v * 2) : (16 + (v - 4) * 2)) + kh;
        afrag[2 * v + 0] = lA[cur][row][kbb + 0];
        afrag[2 * v + 1] = lA[cur][row][kbb + 1];
      }
    }
    // --- 8 B fragments, double-buffered against the WMMAs ---
    auto loadB = [&](int tt) {
      v16bf bf;
      const int aL = nL + (tt << 4);
#pragma unroll
      for (int e = 0; e < 16; ++e) bf[e] = lB[cur][aL][(kb >> 1) + e + ((e & 8) ? 8 : 0)];
      return bf;
    };
    v16bf bcur = loadB(0);
#pragma unroll
    for (int tt = 0; tt < 8; ++tt) {
      v16bf bnext = (tt < 7) ? loadB(tt + 1) : bcur;
      acc[tt] = __builtin_amdgcn_wmma_f32_16x16x32_bf16(
          false, afrag, false, bcur, (short)0, acc[tt], false, false);
      bcur = bnext;
    }

    if (it < 31)   // convert + stage next A tile after compute (latency hidden)
      *(v4bf*)&lA[nxt][aRow][dv * 4] = f2bf4(areg);
    __syncthreads();   // reads of `cur` done before it is overwritten
  }

  // ---------------- fused epilogue ----------------
  const int halfM = ((lane >> 4) & 1) << 3;   // C layout: M = r + 8*(lane/16)
  float lap[8], lac[8], lan[8];
#pragma unroll
  for (int r = 0; r < 8; ++r) {
    const int k = k0 + (waveM << 4) + halfM + r;
    const size_t base = (size_t)b * KLEN + k;
    lac[r] = la[base];
    lap[r] = (k > 0)        ? la[base - 1] : 0.f;
    lan[r] = (k < KLEN - 1) ? la[base + 1] : 0.f;
  }
  float rowsum[8];
#pragma unroll
  for (int r = 0; r < 8; ++r) rowsum[r] = 0.f;

#pragma unroll
  for (int tt = 0; tt < 8; ++tt) {
    const int a   = (waveN << 7) + (tt << 4) + (lane & 15);
    const float qba = qb[b * ATT + a];
    const float cw0 = conv_w[a * 3 + 0];
    const float cw1 = conv_w[a * 3 + 1];
    const float cw2 = conv_w[a * 3 + 2];
    const float cb  = conv_b[a];
    const float fw  = fc_w[a];
#pragma unroll
    for (int r = 0; r < 8; ++r) {
      const float e = fast_tanh(acc[tt][r] + qba +
                                lap[r] * cw0 + lac[r] * cw1 + lan[r] * cw2 + cb);
      rowsum[r] += e * fw;
    }
  }
  // reduce over N (16 lanes per half keep distinct M)
#pragma unroll
  for (int r = 0; r < 8; ++r) {
    float v = rowsum[r];
    v += __shfl_xor(v, 1); v += __shfl_xor(v, 2);
    v += __shfl_xor(v, 4); v += __shfl_xor(v, 8);
    if ((lane & 15) == 0)
      atomicAdd(&sc[(waveM << 4) + halfM + r], v);   // ds_add_f32
  }
  __syncthreads();
  if (t < 32) scores[m0 + t] = sc[t] + fc_b[0];
}

// ------------------------------ kernel 3 -----------------------------------
__global__ __launch_bounds__(256) void softmax_kernel(
    const float* __restrict__ scores, float* __restrict__ attn) {
  __shared__ float redm[8];
  __shared__ float reds[8];
  const int b = blockIdx.x, t = threadIdx.x;
  const int lane = t & 31, wave = t >> 5;
  const float* s = scores + (size_t)b * KLEN;
  float vals[8];
  float m = -INFINITY;
#pragma unroll
  for (int j = 0; j < 8; ++j) { vals[j] = s[t + j * 256]; m = fmaxf(m, vals[j]); }
  for (int o = 1; o < 32; o <<= 1) m = fmaxf(m, __shfl_xor(m, o));
  if (lane == 0) redm[wave] = m;
  __syncthreads();
  float bm = redm[0];
#pragma unroll
  for (int w = 1; w < 8; ++w) bm = fmaxf(bm, redm[w]);
  float sum = 0.f;
#pragma unroll
  for (int j = 0; j < 8; ++j) { vals[j] = __expf(vals[j] - bm); sum += vals[j]; }
  for (int o = 1; o < 32; o <<= 1) sum += __shfl_xor(sum, o);
  if (lane == 0) reds[wave] = sum;
  __syncthreads();
  float bs = 0.f;
#pragma unroll
  for (int w = 0; w < 8; ++w) bs += reds[w];
  const float inv = 1.f / bs;
#pragma unroll
  for (int j = 0; j < 8; ++j) attn[(size_t)b * KLEN + t + j * 256] = vals[j] * inv;
}

// ------------------------------ kernel 4 -----------------------------------
__global__ __launch_bounds__(256) void zero_ctx_kernel(float* __restrict__ ctx) {
  ctx[blockIdx.x * 256 + threadIdx.x] = 0.f;
}

__global__ __launch_bounds__(256) void context_kernel(
    const float* __restrict__ attn, const float* __restrict__ value,
    float* __restrict__ ctx) {
  __shared__ float sa[256];
  const int bx = blockIdx.x;
  const int b  = bx >> 5;
  const int rem = bx & 31;
  const int kc = rem >> 2;           // 8 k-chunks of 256
  const int dc = rem & 3;            // 4 d-chunks of 256
  const int d  = dc * 256 + threadIdx.x;
  const int kbeg = kc * 256;
  sa[threadIdx.x] = attn[(size_t)b * KLEN + kbeg + threadIdx.x];
  __syncthreads();
  const float* vb = value + ((size_t)b * KLEN + kbeg) * DEC;
  float accv = 0.f;
#pragma unroll 4
  for (int kk = 0; kk < 256; ++kk)
    accv += sa[kk] * vb[(size_t)kk * DEC + d];
  atomicAdd(&ctx[(size_t)b * DEC + d], accv);   // global_atomic_add_f32
}

// ---------------------------------------------------------------------------
extern "C" void kernel_launch(void* const* d_in, const int* in_sizes, int n_in,
                              void* d_out, int out_size, void* d_ws, size_t ws_size,
                              hipStream_t stream) {
  const float* query  = (const float*)d_in[0];
  const float* keyf   = (const float*)d_in[1];
  const float* value  = (const float*)d_in[2];
  const float* la     = (const float*)d_in[3];
  const float* conv_w = (const float*)d_in[4];
  const float* conv_b = (const float*)d_in[5];
  const float* Wq     = (const float*)d_in[6];
  const float* Wk     = (const float*)d_in[7];
  const float* bias   = (const float*)d_in[8];
  const float* fc_w   = (const float*)d_in[9];
  const float* fc_b   = (const float*)d_in[10];

  float* out  = (float*)d_out;
  float* ctx  = out;                 // context: B*DEC = 32768 floats
  float* attn = out + B_ * DEC;      // attn:    B*KLEN = 65536 floats

  float*  qb     = (float*)d_ws;                  // 16384 floats
  float*  scores = qb + B_ * ATT;                 // 65536 floats
  __bf16* wkbf   = (__bf16*)(scores + B_ * KLEN); // 512*1024 bf16 (1 MB)

  convert_wk_kernel<<<(ATT * DEC) / (256 * 4), 256, 0, stream>>>(Wk, wkbf);
  qproj_kernel<<<(B_ * ATT) / 8, 256, 0, stream>>>(query, Wq, bias, qb);
  zero_ctx_kernel<<<(B_ * DEC) / 256, 256, 0, stream>>>(ctx);
  scores_wmma_kernel<<<(B_ * KLEN) / 32, 256, 0, stream>>>(
      keyf, wkbf, la, qb, conv_w, conv_b, fc_w, fc_b, scores);
  softmax_kernel<<<B_, 256, 0, stream>>>(scores, attn);
  context_kernel<<<B_ * 32, 256, 0, stream>>>(attn, value, ctx);
}
